// SeqGGNN_53008486367767
// MI455X (gfx1250) — compile-verified, matrix operand order checked by reading, not run on hardware
//
#include <hip/hip_runtime.h>
#include <hip/hip_bf16.h>
#include <math.h>

// ---------------------------------------------------------------------------
// SeqGGNN forward on MI455X (gfx1250), bf16 WMMA pipeline, f32 accumulation.
// ---------------------------------------------------------------------------

typedef __bf16 bf16;
typedef __attribute__((ext_vector_type(16))) bf16  v16bf;
typedef __attribute__((ext_vector_type(8)))  bf16  v8bf;
typedef __attribute__((ext_vector_type(8)))  float v8f;

#define BATCH 128
#define SEQL  511
#define NTOK  (BATCH * SEQL)   // 65408, divisible by 16 -> 4088 token tiles
#define HID   256
#define GATE  768
#define VOC   32000
#define EMB   128
#define NSTEPS 10

// --- WMMA wrapper ----------------------------------------------------------
__device__ __forceinline__ v8f wmma_bf16(v16bf a, v16bf b, v8f c) {
  // (neg_a, A, neg_b, B, c_mod, C, reuse_a, reuse_b)
  return __builtin_amdgcn_wmma_f32_16x16x32_bf16(false, a, false, b, (short)0, c,
                                                 false, false);
}

// Load one lane's slice of a 16x32 (MxK) bf16 fragment from a row-major
// matrix.  `p` must already point at (row(lane) * ld + kh) where
// kh = (lane&16)?8:0; pass k0 (multiple of 32) as offset.
// Elements 0..7  <- K = kh + k0 + 0..7
// Elements 8..15 <- K = kh + k0 + 16..23
__device__ __forceinline__ v16bf frag_ld(const bf16* __restrict__ p, int k0) {
  v8bf lo = *(const v8bf*)(p + k0);
  v8bf hi = *(const v8bf*)(p + k0 + 16);
  v16bf f;
#pragma unroll
  for (int i = 0; i < 8; ++i) { f[i] = lo[i]; f[i + 8] = hi[i]; }
  return f;
}

__device__ __forceinline__ float sigmoidf_(float x) {
  return 1.0f / (1.0f + expf(-x));
}

// --- weight conversion -----------------------------------------------------
__global__ void cvt_f32_bf16_kernel(const float* __restrict__ src,
                                    bf16* __restrict__ dst, int n) {
  int i = blockIdx.x * blockDim.x + threadIdx.x;
  if (i < n) dst[i] = (bf16)src[i];
}

// h0[t, c] = c<128 ? emb_w[x[t], c] : posemb_w[l+1, c-128]
__global__ void embed_kernel(const int* __restrict__ x,
                             const float* __restrict__ emb_w,
                             const float* __restrict__ posemb_w,
                             bf16* __restrict__ h0) {
  int idx = blockIdx.x * blockDim.x + threadIdx.x;
  if (idx >= NTOK * HID) return;
  int t = idx >> 8;
  int c = idx & 255;
  int l = t % SEQL;
  float v;
  if (c < EMB) v = emb_w[(size_t)x[t] * EMB + c];
  else         v = posemb_w[(size_t)(l + 1) * EMB + (c - EMB)];
  h0[idx] = (bf16)v;
}

// --- per-step kernel 1: m1 = h*W1^T, m2 = h*W2^T ---------------------------
// grid (4088, 4), block 128 (4 waves); wave w handles j-tile blockIdx.y*4+w.
__global__ void mm12_kernel(const bf16* __restrict__ h,
                            const bf16* __restrict__ w1b,
                            const bf16* __restrict__ w2b,
                            bf16* __restrict__ m1,
                            bf16* __restrict__ m2) {
  const int wave  = threadIdx.x >> 5;
  const int lane  = threadIdx.x & 31;
  const int t0    = blockIdx.x * 16;
  const int jtile = blockIdx.y * 4 + wave;      // 0..15
  const int j0    = jtile * 16;

  const int r  = lane & 15;
  const int kh = (lane & 16) ? 8 : 0;

  const bf16* pa  = h   + (size_t)(t0 + r) * HID + kh;   // A row = token
  const bf16* pb1 = w1b + (size_t)(j0 + r) * HID + kh;   // B col = out row
  const bf16* pb2 = w2b + (size_t)(j0 + r) * HID + kh;

  v8f c1 = {}, c2 = {};
#pragma unroll
  for (int k0 = 0; k0 < HID; k0 += 32) {
    v16bf fa = frag_ld(pa, k0);
    v16bf b1 = frag_ld(pb1, k0);
    v16bf b2 = frag_ld(pb2, k0);
    c1 = wmma_bf16(fa, b1, c1);
    c2 = wmma_bf16(fa, b2, c2);
  }

  const int ncol = j0 + r;
  const int mo   = (lane & 16) ? 8 : 0;
#pragma unroll
  for (int rr = 0; rr < 8; ++rr) {
    size_t o = (size_t)(t0 + rr + mo) * HID + ncol;
    m1[o] = (bf16)c1[rr];
    m2[o] = (bf16)c2[rr];
  }
}

// --- per-step kernel 2: a = shift_down(m1) + shift_up(m2) ------------------
// Pure bandwidth pass (≈100 MB/step vs 23.3 TB/s — negligible); keeps the
// GRU GEMM inner loop free of elementwise VALU work.
__global__ void shift_add_kernel(const bf16* __restrict__ m1,
                                 const bf16* __restrict__ m2,
                                 bf16* __restrict__ a) {
  int i = blockIdx.x * blockDim.x + threadIdx.x;   // 8-element chunks
  if (i >= NTOK * (HID / 8)) return;
  const int t = i >> 5;          // 32 chunks per token row
  const int l = t % SEQL;
  const size_t off = (size_t)i * 8;

  float acc[8];
#pragma unroll
  for (int j = 0; j < 8; ++j) acc[j] = 0.0f;
  if (l != 0) {
    v8bf v = *(const v8bf*)(m1 + off - HID);
#pragma unroll
    for (int j = 0; j < 8; ++j) acc[j] += (float)v[j];
  }
  if (l != SEQL - 1) {
    v8bf v = *(const v8bf*)(m2 + off + HID);
#pragma unroll
    for (int j = 0; j < 8; ++j) acc[j] += (float)v[j];
  }
  v8bf r;
#pragma unroll
  for (int j = 0; j < 8; ++j) r[j] = (bf16)acc[j];
  *(v8bf*)(a + off) = r;
}

// --- per-step kernel 3: fused GRU ------------------------------------------
// gi = a*Wi^T + bi ; gh = h*Wh^T + bh ; GRU update -> hnext
// grid (4088, 4), block 128; wave handles one (16-token x 16-hidden) tile,
// holding 6 f32 accumulators (r/z/n gates for gi and gh).
__global__ void gru_step_kernel(const bf16* __restrict__ hcur,
                                const bf16* __restrict__ abuf,
                                const bf16* __restrict__ wib,
                                const bf16* __restrict__ whb,
                                const float* __restrict__ bi,
                                const float* __restrict__ bh,
                                bf16* __restrict__ hnext) {
  const int wave  = threadIdx.x >> 5;
  const int lane  = threadIdx.x & 31;
  const int t0    = blockIdx.x * 16;
  const int jtile = blockIdx.y * 4 + wave;      // 0..15
  const int j0    = jtile * 16;

  const int r  = lane & 15;
  const int kh = (lane & 16) ? 8 : 0;

  const int t = t0 + r;
  const bf16* pa = abuf + (size_t)t * HID + kh;
  const bf16* ph = hcur + (size_t)t * HID + kh;

  // B-side lane columns: Wi/Wh rows (gate g block + j0 + lane%16).
  const bf16* pir = wib + (size_t)(0 * HID + j0 + r) * HID + kh;
  const bf16* piz = wib + (size_t)(1 * HID + j0 + r) * HID + kh;
  const bf16* pin = wib + (size_t)(2 * HID + j0 + r) * HID + kh;
  const bf16* phr = whb + (size_t)(0 * HID + j0 + r) * HID + kh;
  const bf16* phz = whb + (size_t)(1 * HID + j0 + r) * HID + kh;
  const bf16* phn = whb + (size_t)(2 * HID + j0 + r) * HID + kh;

  v8f cir = {}, ciz = {}, cin = {}, chr_ = {}, chz = {}, chn = {};
#pragma unroll
  for (int k0 = 0; k0 < HID; k0 += 32) {
    v16bf fa = frag_ld(pa, k0);
    v16bf fh = frag_ld(ph, k0);

    v16bf bir = frag_ld(pir, k0);
    v16bf biz = frag_ld(piz, k0);
    v16bf bin = frag_ld(pin, k0);
    v16bf bhr = frag_ld(phr, k0);
    v16bf bhz = frag_ld(phz, k0);
    v16bf bhn = frag_ld(phn, k0);

    cir  = wmma_bf16(fa, bir, cir);
    ciz  = wmma_bf16(fa, biz, ciz);
    cin  = wmma_bf16(fa, bin, cin);
    chr_ = wmma_bf16(fh, bhr, chr_);
    chz  = wmma_bf16(fh, bhz, chz);
    chn  = wmma_bf16(fh, bhn, chn);
  }

  // Epilogue: C/D layout -> VGPR rr = row rr + 8*(lane>=16), col = lane%16.
  const int ncol = j0 + r;
  const float bir_ = bi[ncol],           bhr_ = bh[ncol];
  const float biz_ = bi[HID + ncol],     bhz_ = bh[HID + ncol];
  const float bin_ = bi[2 * HID + ncol], bhn_ = bh[2 * HID + ncol];
  const int mo = (lane & 16) ? 8 : 0;
#pragma unroll
  for (int rr = 0; rr < 8; ++rr) {
    const size_t to = (size_t)(t0 + rr + mo) * HID + ncol;
    float hold = (float)hcur[to];
    float rg = sigmoidf_(cir[rr] + bir_ + chr_[rr] + bhr_);
    float zg = sigmoidf_(ciz[rr] + biz_ + chz[rr] + bhz_);
    float ng = tanhf(cin[rr] + bin_ + rg * (chn[rr] + bhn_));
    hnext[to] = (bf16)((1.0f - zg) * ng + zg * hold);
  }
}

// --- final projection: out = h[:, L-1, :] @ Wout^T + bout ------------------
// grid (8, 500), block 128; wave handles (16-batch x 16-vocab) tile.
__global__ void out_kernel(const bf16* __restrict__ hfin,
                           const bf16* __restrict__ woutb,
                           const float* __restrict__ bout,
                           float* __restrict__ out) {
  const int wave  = threadIdx.x >> 5;
  const int lane  = threadIdx.x & 31;
  const int mtile = blockIdx.x;                 // 0..7   (128 batch rows)
  const int ntile = blockIdx.y * 4 + wave;      // 0..1999 (32000 vocab)

  const int r  = lane & 15;
  const int kh = (lane & 16) ? 8 : 0;

  const int b = mtile * 16 + r;                 // batch row for A lane
  const bf16* pa = hfin + ((size_t)b * SEQL + (SEQL - 1)) * HID + kh;
  const bf16* pb = woutb + (size_t)(ntile * 16 + r) * HID + kh;

  v8f acc = {};
#pragma unroll
  for (int k0 = 0; k0 < HID; k0 += 32) {
    v16bf fa = frag_ld(pa, k0);
    v16bf fb = frag_ld(pb, k0);
    acc = wmma_bf16(fa, fb, acc);
  }

  const int ncol = ntile * 16 + r;
  const float bo = bout[ncol];
  const int mo = (lane & 16) ? 8 : 0;
#pragma unroll
  for (int rr = 0; rr < 8; ++rr) {
    int bb = mtile * 16 + rr + mo;
    out[(size_t)bb * VOC + ncol] = acc[rr] + bo;
  }
}

// ---------------------------------------------------------------------------
extern "C" void kernel_launch(void* const* d_in, const int* in_sizes, int n_in,
                              void* d_out, int out_size, void* d_ws, size_t ws_size,
                              hipStream_t stream) {
  const int*   x        = (const int*)d_in[0];
  const float* emb_w    = (const float*)d_in[1];
  const float* posemb_w = (const float*)d_in[2];
  const float* W1       = (const float*)d_in[3];
  const float* W2       = (const float*)d_in[4];
  const float* Wi       = (const float*)d_in[5];
  const float* Wh       = (const float*)d_in[6];
  const float* bi       = (const float*)d_in[7];
  const float* bh       = (const float*)d_in[8];
  const float* Wout     = (const float*)d_in[9];
  const float* bout     = (const float*)d_in[10];
  float* out = (float*)d_out;

  // workspace carve-out (all 256B aligned)
  char* ws = (char*)d_ws;
  size_t off = 0;
  auto carve = [&](size_t bytes) -> void* {
    void* p = ws + off;
    off += (bytes + 255) & ~(size_t)255;
    return p;
  };
  const size_t hbytes = (size_t)NTOK * HID * sizeof(bf16);   // 33.5 MB
  bf16* hA   = (bf16*)carve(hbytes);
  bf16* hB   = (bf16*)carve(hbytes);
  bf16* m1   = (bf16*)carve(hbytes);
  bf16* m2   = (bf16*)carve(hbytes);
  bf16* abuf = (bf16*)carve(hbytes);
  bf16* w1b  = (bf16*)carve((size_t)HID * HID * sizeof(bf16));
  bf16* w2b  = (bf16*)carve((size_t)HID * HID * sizeof(bf16));
  bf16* wib  = (bf16*)carve((size_t)GATE * HID * sizeof(bf16));
  bf16* whb  = (bf16*)carve((size_t)GATE * HID * sizeof(bf16));
  bf16* wout = (bf16*)carve((size_t)VOC * HID * sizeof(bf16));

  // 1) weight conversion to bf16
  {
    int n;
    n = HID * HID;
    cvt_f32_bf16_kernel<<<(n + 255) / 256, 256, 0, stream>>>(W1, w1b, n);
    cvt_f32_bf16_kernel<<<(n + 255) / 256, 256, 0, stream>>>(W2, w2b, n);
    n = GATE * HID;
    cvt_f32_bf16_kernel<<<(n + 255) / 256, 256, 0, stream>>>(Wi, wib, n);
    cvt_f32_bf16_kernel<<<(n + 255) / 256, 256, 0, stream>>>(Wh, whb, n);
    n = VOC * HID;
    cvt_f32_bf16_kernel<<<(n + 255) / 256, 256, 0, stream>>>(Wout, wout, n);
  }

  // 2) embedding -> h0 (bf16)
  {
    int n = NTOK * HID;
    embed_kernel<<<(n + 255) / 256, 256, 0, stream>>>(x, emb_w, posemb_w, hA);
  }

  // 3) 10 scan steps, ping-pong h buffers
  bf16* hc = hA;
  bf16* hn = hB;
  const dim3 grid(NTOK / 16, 4);   // 4088 token tiles x (4 blocks * 4 waves = 16 j-tiles)
  const int nchunks = NTOK * (HID / 8);
  for (int s = 0; s < NSTEPS; ++s) {
    mm12_kernel<<<grid, 128, 0, stream>>>(hc, w1b, w2b, m1, m2);
    shift_add_kernel<<<(nchunks + 255) / 256, 256, 0, stream>>>(m1, m2, abuf);
    gru_step_kernel<<<grid, 128, 0, stream>>>(hc, abuf, wib, whb, bi, bh, hn);
    bf16* tmp = hc; hc = hn; hn = tmp;
  }

  // 4) output projection
  out_kernel<<<dim3(8, VOC / 64), 128, 0, stream>>>(hc, wout, bout, out);

  (void)in_sizes; (void)n_in; (void)out_size; (void)ws_size;
}
